// STGraphConstructor_54829552501064
// MI455X (gfx1250) — matrix operand order repaired and checked by reading, not compiled, not run on hardware
//
#include <hip/hip_runtime.h>
#include <hip/hip_bf16.h>
#include <math.h>

typedef _Float16 v16h __attribute__((ext_vector_type(16)));
typedef float    v8f  __attribute__((ext_vector_type(8)));

#define BDIM  4
#define NSP   5000
#define TTM   96
#define DDIM  64
#define MTOT  (NSP + TTM)                 /* 5096 */
#define TILES 319                          /* ceil(5096/16) */
#define CPW   4                            /* column tiles per wave (16x64 strip) */
#define CGRP  ((TILES + CPW - 1) / CPW)    /* 80 */

// Row pointer into the virtual concatenated [B, M, D] tensor.
// Out-of-range rows are clamped to row 0: they only feed output rows/cols >= M,
// which are never stored, so garbage there is harmless. This keeps every load
// unconditional (no exec-mask divergence, no zero-fill mov walls).
__device__ __forceinline__ const float* row_ptr(const float* __restrict__ sp,
                                                const float* __restrict__ tp,
                                                int b, int r) {
    if (r >= MTOT) r = 0;
    if (r < NSP)  return sp + ((size_t)b * NSP + r) * DDIM;
    return tp + ((size_t)b * TTM + (r - NSP)) * DDIM;
}

// Load 8 consecutive f32 (two b128s) and convert into v16h elements [base, base+8).
__device__ __forceinline__ void load8h(v16h& a, int base, const float* __restrict__ p) {
    float4 u = ((const float4*)p)[0];
    float4 v = ((const float4*)p)[1];
    a[base + 0] = (_Float16)u.x; a[base + 1] = (_Float16)u.y;
    a[base + 2] = (_Float16)u.z; a[base + 3] = (_Float16)u.w;
    a[base + 4] = (_Float16)v.x; a[base + 5] = (_Float16)v.y;
    a[base + 6] = (_Float16)v.z; a[base + 7] = (_Float16)v.w;
}

// Hardware tanh (gfx1250 V_TANH_F32 trans op). Builtin if available (compiler
// schedules the trans-op hazard); otherwise inline asm + v_nop per ISA rule
// for the trans-op result hazard (a gap before a dependent use).
__device__ __forceinline__ float fast_tanh(float x) {
#if __has_builtin(__builtin_amdgcn_tanhf)
    return __builtin_amdgcn_tanhf(x);
#else
    float r;
    asm volatile("v_tanh_f32 %0, %1\n\tv_nop\n\tv_nop" : "=v"(r) : "v"(x));
    return r;
#endif
}

__global__ __launch_bounds__(256)
void stgc_gram_wmma(const float* __restrict__ sp,
                    const float* __restrict__ tp,
                    float* __restrict__ out) {
    const int lane = threadIdx.x & 31;
    const int wid  = blockIdx.x * (blockDim.x >> 5) + (threadIdx.x >> 5);
    const int tasksPerBatch = TILES * CGRP;
    if (wid >= BDIM * tasksPerBatch) return;   // whole-wave exit only

    const int b       = wid / tasksPerBatch;
    const int rem     = wid % tasksPerBatch;
    const int rowTile = rem / CGRP;
    const int colGrp  = rem % CGRP;

    const int g   = lane >> 4;   // half-wave id
    const int l16 = lane & 15;

    // Per-lane row pointers: A-operand row, and 4 B-operand (column-tile) rows.
    const float* aptr = row_ptr(sp, tp, b, rowTile * 16 + l16);
    const float* bptr[CPW];
#pragma unroll
    for (int j = 0; j < CPW; ++j)
        bptr[j] = row_ptr(sp, tp, b, (colGrp * CPW + j) * 16 + l16);

    v8f acc[CPW] = {};   // f32 accumulators for 4 16x16 tiles

#pragma unroll
    for (int kb = 0; kb < 2; ++kb) {          // D=64 -> two K=32 steps
        // A operand: lane holds (M=l16, K = kb*32 + {g*8..+7, 16+g*8..+7})
        v16h a;
        load8h(a, 0, aptr + kb * 32 + g * 8);
        load8h(a, 8, aptr + kb * 32 + 16 + g * 8);
#pragma unroll
        for (int j = 0; j < CPW; ++j) {
            // B operand (= X_tile^T): lane holds (N=l16, K = kb*32 + {g*8..+7, 16+g*8..+7})
            v16h bm;
            load8h(bm, 0, bptr[j] + kb * 32 + g * 8);
            load8h(bm, 8, bptr[j] + kb * 32 + 16 + g * 8);
            acc[j] = __builtin_amdgcn_wmma_f32_16x16x32_f16(
                         /*neg_a=*/false, a, /*neg_b=*/false, bm,
                         /*c_mod=*/(short)0, acc[j],
                         /*reuse_a=*/false, /*reuse_b=*/false);
        }
    }

    // Epilogue: tanh(relu(gram) + I), streamed with non-temporal stores.
#pragma unroll
    for (int j = 0; j < CPW; ++j) {
        const int col = (colGrp * CPW + j) * 16 + l16;
        if (col >= MTOT) continue;
#pragma unroll
        for (int i = 0; i < 8; ++i) {
            const int row = rowTile * 16 + g * 8 + i;   // C layout: VGPR i, half-wave g
            if (row < MTOT) {
                float v = fmaxf(acc[j][i], 0.0f);
                if (row == col) v += 1.0f;
                v = fast_tanh(v);
                __builtin_nontemporal_store(
                    v, out + ((size_t)b * MTOT + row) * MTOT + col);
            }
        }
    }
}

extern "C" void kernel_launch(void* const* d_in, const int* in_sizes, int n_in,
                              void* d_out, int out_size, void* d_ws, size_t ws_size,
                              hipStream_t stream) {
    (void)in_sizes; (void)n_in; (void)out_size; (void)d_ws; (void)ws_size;
    const float* spatial  = (const float*)d_in[0];   // [4, 5000, 64] f32
    const float* temporal = (const float*)d_in[1];   // [4,   96, 64] f32
    float* out = (float*)d_out;                      // [4, 5096, 5096] f32

    const int totalWaves    = BDIM * TILES * CGRP;   // 102,080
    const int wavesPerBlock = 8;                     // 256 threads, wave32
    const int blocks = (totalWaves + wavesPerBlock - 1) / wavesPerBlock; // 12,760

    stgc_gram_wmma<<<blocks, 256, 0, stream>>>(spatial, temporal, out);
}
